// ScaledDotAttention_146028888336
// MI455X (gfx1250) — matrix operand order repaired and checked by reading, not compiled, not run on hardware
//
#include <hip/hip_runtime.h>
#include <hip/hip_bf16.h>

// ScaledDotAttention on MI455X (gfx1250), bf16 WMMA path.
// B=4, S=2048 (== query count K), H=1024.

#define DIM_H 1024
#define DIM_B 4
#define DIM_S 2048

typedef __attribute__((ext_vector_type(16))) __bf16 v16bf;
typedef __attribute__((ext_vector_type(8)))  __bf16 bf16x8;
typedef __attribute__((ext_vector_type(8)))  float  v8f;
typedef __attribute__((ext_vector_type(4)))  float  f32x4;

// ---------------------------------------------------------------------------
// Fragment loaders.
// A-matrix 16x32 bf16 layout (ISA 7.12.2): lane = M (lane&15);
//   lanes 0-15 hold K = {kOff..kOff+7, kOff+16..kOff+23} with kOff=0,
//   lanes 16-31 the same with kOff=8.
// ---------------------------------------------------------------------------
__device__ inline v16bf load_fragA(const __bf16* p) {
  bf16x8 lo = *reinterpret_cast<const bf16x8*>(p);
  bf16x8 hi = *reinterpret_cast<const bf16x8*>(p + 16);
  v16bf r;
#pragma unroll
  for (int i = 0; i < 8; ++i) { r[i] = lo[i]; r[i + 8] = hi[i]; }
  return r;
}

__device__ inline v16bf load_fragA(const float* p) {
  // fp32 source: load 2x(8 floats) and convert to bf16 in-register
  // (v_cvt_pk_bf16_f32 co-executes with the XDL WMMA pipe).
  f32x4 a0 = *reinterpret_cast<const f32x4*>(p);
  f32x4 a1 = *reinterpret_cast<const f32x4*>(p + 4);
  f32x4 b0 = *reinterpret_cast<const f32x4*>(p + 16);
  f32x4 b1 = *reinterpret_cast<const f32x4*>(p + 20);
  v16bf r;
#pragma unroll
  for (int i = 0; i < 4; ++i) {
    r[i]      = (__bf16)a0[i];
    r[4 + i]  = (__bf16)a1[i];
    r[8 + i]  = (__bf16)b0[i];
    r[12 + i] = (__bf16)b1[i];
  }
  return r;
}

// B-matrix 32x16 bf16 layout: lane = N (lane&15); same K interleave as A:
// lanes 0-15 hold K = {0..7, 16..23}, lanes 16-31 hold K = {8..15, 24..31}.
// Source row-major [N][K] -> 2x8 bf16 per lane.
__device__ inline v16bf load_fragB(const __bf16* p) {
  bf16x8 lo = *reinterpret_cast<const bf16x8*>(p);
  bf16x8 hi = *reinterpret_cast<const bf16x8*>(p + 16);
  v16bf r;
#pragma unroll
  for (int i = 0; i < 8; ++i) { r[i] = lo[i]; r[i + 8] = hi[i]; }
  return r;
}

// ---------------------------------------------------------------------------
// Unified NT-GEMM:  C[M,N] = scale * (A[MxK] . B[NxK]^T) (+ bias[N])
// Both operands row-major with contiguous reduction dim.
// 256 threads = 8 waves; wave tile 32x64 (2 A-frags x 4 B-frags, 8 WMMAs per
// K-step), block tile 64x256.  Optionally stores C transposed (for v^T).
// ---------------------------------------------------------------------------
template <typename TA, typename TO, bool TRANS, bool BIAS>
__global__ __launch_bounds__(256) void gemm_nt_wmma(
    const TA* __restrict__ A, const __bf16* __restrict__ Bm,
    const float* __restrict__ bias, TO* __restrict__ C,
    int K, int lda, int ldb, int ldc,
    long strideA, long strideB, long strideC, float scale)
{
  const int lane = threadIdx.x & 31;
  const int wave = threadIdx.x >> 5;
  const int M0 = blockIdx.x * 64  + (wave >> 2) * 32;
  const int N0 = blockIdx.y * 256 + (wave & 3)  * 64;
  const int b  = blockIdx.z;

  const TA*     Ab = A  + (size_t)b * strideA;
  const __bf16* Bb = Bm + (size_t)b * strideB;
  TO*           Cb = C  + (size_t)b * strideC;

  const int r16   = lane & 15;
  const int kOffA = (lane >> 4) << 3;   // 0 or 8   (A interleave)
  const int kOffB = (lane >> 4) << 3;   // 0 or 8   (B interleave, matches A)

  const TA* pa[2];
  pa[0] = Ab + (size_t)(M0 + r16) * lda + kOffA;
  pa[1] = pa[0] + (size_t)16 * lda;
  const __bf16* pb[4];
  pb[0] = Bb + (size_t)(N0 + r16) * ldb + kOffB;
#pragma unroll
  for (int j = 1; j < 4; ++j) pb[j] = pb[0] + (size_t)(16 * j) * ldb;

  v8f acc[2][4] = {};

  for (int k0 = 0; k0 < K; k0 += 32) {
    v16bf af[2], bf[4];
#pragma unroll
    for (int i = 0; i < 2; ++i) af[i] = load_fragA(pa[i] + k0);
#pragma unroll
    for (int j = 0; j < 4; ++j) bf[j] = load_fragB(pb[j] + k0);
#pragma unroll
    for (int i = 0; i < 2; ++i)
#pragma unroll
      for (int j = 0; j < 4; ++j)
        acc[i][j] = __builtin_amdgcn_wmma_f32_16x16x32_bf16(
            false, af[i], false, bf[j], (short)0, acc[i][j], false, false);
  }

  // C/D layout: VGPR r, lanes 0-15 -> M=r, lanes 16-31 -> M=r+8; N = lane&15.
  const int mSub = (lane >> 4) << 3;
  const int nSub = lane & 15;

  float bj[4] = {0.f, 0.f, 0.f, 0.f};
  if (BIAS) {
#pragma unroll
    for (int j = 0; j < 4; ++j) bj[j] = bias[N0 + 16 * j + nSub];
  }

#pragma unroll
  for (int r = 0; r < 8; ++r) {
#pragma unroll
    for (int i = 0; i < 2; ++i) {
      const int m = M0 + 16 * i + mSub + r;
#pragma unroll
      for (int j = 0; j < 4; ++j) {
        const int n = N0 + 16 * j + nSub;
        const float v = acc[i][j][r] * scale + bj[j];
        if (TRANS) Cb[(size_t)n * ldc + m] = (TO)v;
        else       Cb[(size_t)m * ldc + n] = (TO)v;
      }
    }
  }
}

// ---------------------------------------------------------------------------
// fp32 -> bf16 conversion (weights)
// ---------------------------------------------------------------------------
__global__ void cvt_f32_bf16(const float* __restrict__ src,
                             __bf16* __restrict__ dst, int n) {
  int i = blockIdx.x * 256 + threadIdx.x;
  if (i < n) dst[i] = (__bf16)src[i];
}

// ---------------------------------------------------------------------------
// Softmax over s (axis=1 of scores[B,S,K]).  One block per 32-column tile:
// lanes map to consecutive kq so every global access is a coalesced 128B
// line.  Per-column max/sum reduced across the 8 row-groups via LDS.
// Pass 3 re-reads (L2-resident), writes normalized fp32 attn in place, and
// emits bf16 attn^T [B][K][S] through a padded LDS transpose tile.
// ---------------------------------------------------------------------------
__global__ __launch_bounds__(256) void softmax_tile(
    float* __restrict__ attn, __bf16* __restrict__ attnT, int S, int Kq)
{
  const int b   = blockIdx.y;
  const int c0  = blockIdx.x * 32;          // first column of tile
  const int tid = threadIdx.x;
  const int col = tid & 31;                 // column within tile
  const int rg  = tid >> 5;                 // row group 0..7
  const int kq  = c0 + col;

  float* base = attn + (size_t)b * S * Kq;

  __shared__ float red[256];
  // 32 x 64 bf16 transpose tile, row stride 66 halfs = 33 dwords
  // (gcd(33,64)==1 -> conflict-free across all 64 LDS banks).
  __shared__ __bf16 tile[32][66];

  // ---- pass 1: column max ----
  float lmax = -3.402823466e38f;
  for (int s = rg; s < S; s += 8)
    lmax = fmaxf(lmax, base[(size_t)s * Kq + kq]);
  red[tid] = lmax;
  __syncthreads();
  if (tid < 32) {
    float m = red[tid];
#pragma unroll
    for (int g = 1; g < 8; ++g) m = fmaxf(m, red[tid + g * 32]);
    red[tid] = m;
  }
  __syncthreads();
  const float mx = red[col];
  __syncthreads();

  // ---- pass 2: column sum of exp ----
  float lsum = 0.f;
  for (int s = rg; s < S; s += 8)
    lsum += __expf(base[(size_t)s * Kq + kq] - mx);
  red[tid] = lsum;
  __syncthreads();
  if (tid < 32) {
    float t = red[tid];
#pragma unroll
    for (int g = 1; g < 8; ++g) t += red[tid + g * 32];
    red[tid] = t;
  }
  __syncthreads();
  const float inv = 1.f / red[col];
  __syncthreads();

  // ---- pass 3: normalize, write attn (coalesced) + attn^T via LDS ----
  __bf16* tbase = attnT + (size_t)b * Kq * S;
  const int orow = tid >> 3;                // 0..31: output row of tile
  const int oe0  = (tid & 7) * 8;           // 0..56: 8-elem chunk in s
  for (int s0 = 0; s0 < S; s0 += 64) {
#pragma unroll
    for (int i = 0; i < 8; ++i) {
      const int s = s0 + rg + i * 8;
      const float p = __expf(base[(size_t)s * Kq + kq] - mx) * inv;
      base[(size_t)s * Kq + kq] = p;
      tile[col][s - s0] = (__bf16)p;
    }
    __syncthreads();
    __bf16* dst = tbase + (size_t)(c0 + orow) * S + s0 + oe0;
#pragma unroll
    for (int e = 0; e < 8; ++e) dst[e] = tile[orow][oe0 + e];
    __syncthreads();
  }
}

// ---------------------------------------------------------------------------
extern "C" void kernel_launch(void* const* d_in, const int* in_sizes, int n_in,
                              void* d_out, int out_size, void* d_ws,
                              size_t ws_size, hipStream_t stream) {
  (void)in_sizes; (void)n_in; (void)out_size; (void)ws_size;

  const float* queries = (const float*)d_in[0];
  const float* keys    = (const float*)d_in[1];
  const float* values  = (const float*)d_in[2];
  const float* Wq      = (const float*)d_in[3];
  const float* bq      = (const float*)d_in[4];
  const float* Wk      = (const float*)d_in[5];
  const float* bk      = (const float*)d_in[6];
  const float* Wv      = (const float*)d_in[7];
  const float* bv      = (const float*)d_in[8];

  const int H = DIM_H, B = DIM_B, S = DIM_S, Kq = DIM_S;

  // Workspace layout (bytes). attn^T aliases q/k projections (dead by then).
  char* ws = (char*)d_ws;
  size_t off = 0;
  __bf16* wq_bf = (__bf16*)(ws + off); off += (size_t)H * H * 2;
  __bf16* wk_bf = (__bf16*)(ws + off); off += (size_t)H * H * 2;
  __bf16* wv_bf = (__bf16*)(ws + off); off += (size_t)H * H * 2;
  __bf16* qb    = (__bf16*)(ws + off); off += (size_t)B * S * H * 2;
  __bf16* kb    = (__bf16*)(ws + off); off += (size_t)B * S * H * 2;
  __bf16* vT    = (__bf16*)(ws + off); off += (size_t)B * H * S * 2;
  __bf16* attnT = qb;  // 32 MB, reuses qb+kb (32 MB) after scores GEMM

  float* ctx  = (float*)d_out;                 // [B, Kq, H]
  float* attn = ctx + (size_t)B * Kq * H;      // [B, S, Kq]

  const dim3 blk(256);

  // 1) weights -> bf16
  {
    const int n = H * H;
    const dim3 g((n + 255) / 256);
    cvt_f32_bf16<<<g, blk, 0, stream>>>(Wq, wq_bf, n);
    cvt_f32_bf16<<<g, blk, 0, stream>>>(Wk, wk_bf, n);
    cvt_f32_bf16<<<g, blk, 0, stream>>>(Wv, wv_bf, n);
  }

  // 2) projections: M=S, N=H, K=H; fp32 x on-the-fly -> bf16, weights bf16
  const dim3 gp(S / 64, H / 256, B);
  gemm_nt_wmma<float, __bf16, false, true><<<gp, blk, 0, stream>>>(
      queries, wq_bf, bq, qb, H, H, H, H,
      (long)S * H, 0L, (long)S * H, 1.0f);
  gemm_nt_wmma<float, __bf16, false, true><<<gp, blk, 0, stream>>>(
      keys, wk_bf, bk, kb, H, H, H, H,
      (long)S * H, 0L, (long)S * H, 1.0f);
  gemm_nt_wmma<float, __bf16, true, true><<<gp, blk, 0, stream>>>(
      values, wv_bf, bv, vT, H, H, H, S,            // store v^T [B][H][S]
      (long)S * H, 0L, (long)H * S, 1.0f);

  // 3) scores = scale * kb . qb^T  -> fp32 into attn region of d_out
  const float scale = 1.0f / 32.0f;  // 1/sqrt(1024)
  const dim3 gs(S / 64, Kq / 256, B);
  gemm_nt_wmma<__bf16, float, false, false><<<gs, blk, 0, stream>>>(
      kb, qb, nullptr, attn, H, H, H, Kq,
      (long)S * H, (long)S * H, (long)S * Kq, scale);

  // 4) softmax over s; also emit bf16 attn^T
  const dim3 gsm(Kq / 32, B);
  softmax_tile<<<gsm, blk, 0, stream>>>(attn, attnT, S, Kq);

  // 5) context = attn^T . v  -> [B, Kq, H] fp32 (A=attnT [KqxS], B=v^T [HxS])
  const dim3 gc(Kq / 64, H / 256, B);
  gemm_nt_wmma<__bf16, float, false, false><<<gc, blk, 0, stream>>>(
      attnT, vT, nullptr, ctx, S, S, S, H,
      (long)Kq * S, (long)H * S, (long)Kq * H, 1.0f);
}
